// swin_tiny_40475771797866
// MI455X (gfx1250) — matrix operand order repaired and checked by reading, not compiled, not run on hardware
//
#include <hip/hip_runtime.h>
#include <math.h>

typedef __bf16 v16bf __attribute__((ext_vector_type(16)));
typedef __bf16 v8bf  __attribute__((ext_vector_type(8)));
typedef float  v8f   __attribute__((ext_vector_type(8)));

static __device__ inline v16bf cat16(v8bf lo, v8bf hi) {
  return __builtin_shufflevector(lo, hi, 0, 1, 2, 3, 4, 5, 6, 7,
                                 8, 9, 10, 11, 12, 13, 14, 15);
}
// A-fragment: 16 contiguous K halves at k0+half*16 (matches frag_b)
static __device__ inline v16bf frag_a(const __bf16* row, int k0, int half) {
  v8bf lo = *(const v8bf*)(row + k0 + half * 16);
  v8bf hi = *(const v8bf*)(row + k0 + half * 16 + 8);
  return cat16(lo, hi);
}
// B-fragment: 16 contiguous K halves at k0+half*16 (same k order as frag_a)
static __device__ inline v16bf frag_b(const __bf16* row, int k0, int half) {
  v8bf lo = *(const v8bf*)(row + k0 + half * 16);
  v8bf hi = *(const v8bf*)(row + k0 + half * 16 + 8);
  return cat16(lo, hi);
}

// ---------------------------------------------------------------- fp32 -> bf16 convert
__global__ void k_cvt_bf16(const float* __restrict__ in, __bf16* __restrict__ out, long n) {
  long i = (long)blockIdx.x * blockDim.x + threadIdx.x;
  if (i < n) out[i] = (__bf16)in[i];
}

// ---------------------------------------------------------------- patch embed
// x: [B,3,224,224] NCHW, w: [96,3,4,4], out: [B*56*56, 96] token-major fp32
__global__ void k_patch_embed(const float* __restrict__ x, const float* __restrict__ w,
                              const float* __restrict__ bias, float* __restrict__ out, int B) {
  long i = (long)blockIdx.x * blockDim.x + threadIdx.x;
  long total = (long)B * 56 * 56 * 96;
  if (i >= total) return;
  int o = (int)(i % 96);
  long t = i / 96;
  int px = (int)(t % 56);
  long t2 = t / 56;
  int py = (int)(t2 % 56);
  int b  = (int)(t2 / 56);
  float acc = bias[o];
  const float* wp = w + (long)o * 48;
  for (int c = 0; c < 3; ++c)
    for (int ky = 0; ky < 4; ++ky) {
      const float* xr = x + ((((long)b * 3 + c) * 224 + (py * 4 + ky)) * 224 + px * 4);
      const float* wr = wp + (c * 4 + ky) * 4;
      #pragma unroll
      for (int kx = 0; kx < 4; ++kx) acc += xr[kx] * wr[kx];
    }
  out[i] = acc;
}

// ---------------------------------------------------------------- layernorm
// one wave per row; writes fp32 (of) and/or bf16 (ob)
__global__ void k_layernorm(const float* __restrict__ in, const float* __restrict__ g,
                            const float* __restrict__ b, float* __restrict__ of,
                            __bf16* __restrict__ ob, int M, int C) {
  int wave = threadIdx.x >> 5, lane = threadIdx.x & 31;
  int row = blockIdx.x * (blockDim.x >> 5) + wave;
  if (row >= M) return;
  const float* r = in + (long)row * C;
  float s = 0.f;
  for (int c = lane; c < C; c += 32) s += r[c];
  #pragma unroll
  for (int m = 16; m >= 1; m >>= 1) s += __shfl_xor(s, m, 32);
  float mean = s / C;
  float v = 0.f;
  for (int c = lane; c < C; c += 32) { float d = r[c] - mean; v += d * d; }
  #pragma unroll
  for (int m = 16; m >= 1; m >>= 1) v += __shfl_xor(v, m, 32);
  float rstd = rsqrtf(v / C + 1e-5f);
  for (int c = lane; c < C; c += 32) {
    float val = (r[c] - mean) * rstd * g[c] + b[c];
    if (of) of[(long)row * C + c] = val;
    if (ob) ob[(long)row * C + c] = (__bf16)val;
  }
}

// ---------------------------------------------------------------- bf16 WMMA GEMM (2x2 blocked)
// acc = A[M,K] @ W[N,K]^T + bias; one wave computes a 32x32 tile (4 WMMA tiles).
// Output: bf16 (OutB) or fp32 (OutF, optional += accumulate). act==1 -> exact GELU.
// Requires M%32==0, N%32==0, K%32==0.
__global__ void k_gemm_bf16(const __bf16* __restrict__ A, const __bf16* __restrict__ W,
                            const float* __restrict__ bias, float* __restrict__ OutF,
                            __bf16* __restrict__ OutB, int M, int N, int K,
                            int act, int accum) {
  int wave = threadIdx.x >> 5, lane = threadIdx.x & 31;
  int tn2cnt = N >> 5;
  long tile = (long)blockIdx.x * (blockDim.x >> 5) + wave;
  long tot = (long)(M >> 5) * tn2cnt;
  if (tile >= tot) return;
  int tm2 = (int)(tile / tn2cnt), tn2 = (int)(tile % tn2cnt);
  int half = lane >> 4, r = lane & 15;

  const __bf16* a0p = A + (long)(tm2 * 32 + r) * K;
  const __bf16* a1p = a0p + (long)16 * K;
  const __bf16* b0p = W + (long)(tn2 * 32 + r) * K;
  const __bf16* b1p = b0p + (long)16 * K;

  v8f acc00 = {0.f,0.f,0.f,0.f,0.f,0.f,0.f,0.f};
  v8f acc01 = acc00, acc10 = acc00, acc11 = acc00;

  for (int k0 = 0; k0 < K; k0 += 32) {
    __builtin_prefetch(a0p + k0 + 64, 0, 0);
    __builtin_prefetch(b0p + k0 + 64, 0, 0);
    v16bf a0 = frag_a(a0p, k0, half);
    v16bf a1 = frag_a(a1p, k0, half);
    v16bf b0 = frag_b(b0p, k0, half);
    v16bf b1 = frag_b(b1p, k0, half);
    acc00 = __builtin_amdgcn_wmma_f32_16x16x32_bf16(false, a0, false, b0, (short)0, acc00, false, false);
    acc01 = __builtin_amdgcn_wmma_f32_16x16x32_bf16(false, a0, false, b1, (short)0, acc01, false, false);
    acc10 = __builtin_amdgcn_wmma_f32_16x16x32_bf16(false, a1, false, b0, (short)0, acc10, false, false);
    acc11 = __builtin_amdgcn_wmma_f32_16x16x32_bf16(false, a1, false, b1, (short)0, acc11, false, false);
  }

  v8f* accs[4] = {&acc00, &acc01, &acc10, &acc11};
  #pragma unroll
  for (int sm = 0; sm < 2; ++sm)
    #pragma unroll
    for (int sn = 0; sn < 2; ++sn) {
      v8f& acc = *accs[sm * 2 + sn];
      int n = tn2 * 32 + sn * 16 + r;
      float bb = bias ? bias[n] : 0.f;
      #pragma unroll
      for (int v = 0; v < 8; ++v) {
        int m = tm2 * 32 + sm * 16 + v + 8 * half;
        float val = acc[v] + bb;
        if (act == 1) val = 0.5f * val * (1.f + erff(val * 0.70710678118654752f));
        long o = (long)m * N + n;
        if (OutB) OutB[o] = (__bf16)val;
        else if (accum) OutF[o] += val;
        else OutF[o] = val;
      }
    }
}

// ---------------------------------------------------------------- window attention
// qkv: bf16 [B*H*W, 3C]; out: bf16 [B*H*W, C]; one wave per (window, head), hd=32
__global__ __launch_bounds__(32) void k_attention(
    const __bf16* __restrict__ qkv, __bf16* __restrict__ out, const float* __restrict__ rpb,
    int B, int H, int W, int C, int heads, int shift) {
  const int hd = 32;
  int lane = threadIdx.x;
  int head = blockIdx.y;
  int nw = W / 7, nh = H / 7;
  int win = blockIdx.x;
  int wx = win % nw;
  int wy = (win / nw) % nh;
  int b  = win / (nw * nh);
  const float scale = 0.17677669529663687f;  // 1/sqrt(32)

  __shared__ int tokS[64];
  __shared__ int regS[64];
  __shared__ __align__(16) __bf16 Qs[64 * 32];
  __shared__ __align__(16) __bf16 Ks[64 * 32];
  __shared__ __align__(16) __bf16 Vs[64 * 32];
  __shared__ __align__(16) __bf16 Ps[64 * 64];

  for (int i = lane; i < 64; i += 32) {
    int tok = -1, reg = 0;
    if (i < 49) {
      int iy = i / 7, ix = i % 7;
      int sy = wy * 7 + iy, sx = wx * 7 + ix;
      int oy = sy + shift; if (oy >= H) oy -= H;
      int ox = sx + shift; if (ox >= W) ox -= W;
      tok = (b * H + oy) * W + ox;
      if (shift > 0) {
        int rh = (sy < H - 7) ? 0 : ((sy < H - shift) ? 1 : 2);
        int rw = (sx < W - 7) ? 0 : ((sx < W - shift) ? 1 : 2);
        reg = rh * 3 + rw;
      }
    }
    tokS[i] = tok; regS[i] = reg;
  }
  __syncthreads();

  int C3 = 3 * C;
  for (int j = 0; j < 64; ++j) {
    int tok = tokS[j];
    int d = lane;
    if (tok >= 0) {
      long base = (long)tok * C3 + head * hd + d;
      Qs[j * 32 + d] = qkv[base];
      Ks[j * 32 + d] = qkv[base + C];
      Vs[j * 32 + d] = qkv[base + 2 * C];
    } else {
      Qs[j * 32 + d] = (__bf16)0.f;
      Ks[j * 32 + d] = (__bf16)0.f;
      Vs[j * 32 + d] = (__bf16)0.f;
    }
  }
  __syncthreads();

  int half = lane >> 4, r = lane & 15;

  // S = (q @ k^T)*scale + bias (+mask), softmax, store P (bf16) to LDS
  for (int ti = 0; ti < 4; ++ti) {
    v16bf aq = frag_a(Qs + (ti * 16 + r) * 32, 0, half);
    v8f s[4];
    #pragma unroll
    for (int tj = 0; tj < 4; ++tj) {
      v16bf bk = frag_b(Ks + (tj * 16 + r) * 32, 0, half);
      v8f z = {0.f,0.f,0.f,0.f,0.f,0.f,0.f,0.f};
      s[tj] = __builtin_amdgcn_wmma_f32_16x16x32_bf16(false, aq, false, bk,
                                                      (short)0, z, false, false);
    }
    #pragma unroll
    for (int v = 0; v < 8; ++v) {
      int n = ti * 16 + v + 8 * half;
      int niy = (n < 49) ? n / 7 : 0;
      int nix = (n < 49) ? n % 7 : 0;
      int nreg = regS[n];
      float x[4];
      float rmax = -1e30f;
      #pragma unroll
      for (int tj = 0; tj < 4; ++tj) {
        int m = tj * 16 + r;
        float val = s[tj][v] * scale;
        if (m < 49 && n < 49) {
          int dy = niy - m / 7 + 6;
          int dx = nix - m % 7 + 6;
          val += rpb[(dy * 13 + dx) * heads + head];
          if (shift > 0 && nreg != regS[m]) val += -100.f;
        }
        if (m >= 49) val = -1e30f;
        x[tj] = val;
        rmax = fmaxf(rmax, val);
      }
      #pragma unroll
      for (int msk = 1; msk < 16; msk <<= 1) rmax = fmaxf(rmax, __shfl_xor(rmax, msk, 32));
      float rsum = 0.f;
      #pragma unroll
      for (int tj = 0; tj < 4; ++tj) { x[tj] = __expf(x[tj] - rmax); rsum += x[tj]; }
      #pragma unroll
      for (int msk = 1; msk < 16; msk <<= 1) rsum += __shfl_xor(rsum, msk, 32);
      float inv = 1.f / rsum;
      #pragma unroll
      for (int tj = 0; tj < 4; ++tj) Ps[n * 64 + tj * 16 + r] = (__bf16)(x[tj] * inv);
    }
  }
  __syncthreads();

  // O = P @ V  (K = 64 over m, N = 32 over d)
  for (int ti = 0; ti < 4; ++ti) {
    v8f o0 = {0.f,0.f,0.f,0.f,0.f,0.f,0.f,0.f};
    v8f o1 = o0;
    #pragma unroll
    for (int kk = 0; kk < 2; ++kk) {
      v16bf ap = frag_a(Ps + (ti * 16 + r) * 64, kk * 32, half);
      v16bf bv;
      #pragma unroll
      for (int i = 0; i < 16; ++i) bv[i] = Vs[(kk * 32 + half * 16 + i) * 32 + r];
      o0 = __builtin_amdgcn_wmma_f32_16x16x32_bf16(false, ap, false, bv, (short)0, o0, false, false);
      #pragma unroll
      for (int i = 0; i < 16; ++i) bv[i] = Vs[(kk * 32 + half * 16 + i) * 32 + 16 + r];
      o1 = __builtin_amdgcn_wmma_f32_16x16x32_bf16(false, ap, false, bv, (short)0, o1, false, false);
    }
    #pragma unroll
    for (int v = 0; v < 8; ++v) {
      int n = ti * 16 + v + 8 * half;
      if (n < 49) {
        long base = (long)tokS[n] * C + head * hd;
        out[base + r]      = (__bf16)o0[v];
        out[base + 16 + r] = (__bf16)o1[v];
      }
    }
  }
}

// ---------------------------------------------------------------- patch merging gather
// in: [B,H,W,C] fp32 -> out: [B,H/2,W/2,4C] fp32
__global__ void k_merge_gather(const float* __restrict__ in, float* __restrict__ out,
                               int B, int H, int W, int C) {
  int H2 = H / 2, W2 = W / 2;
  long i = (long)blockIdx.x * blockDim.x + threadIdx.x;
  long total = (long)B * H2 * W2 * 4 * C;
  if (i >= total) return;
  int c4 = (int)(i % (4 * C));
  long t = i / (4 * C);
  int x2 = (int)(t % W2);
  long t2 = t / W2;
  int y2 = (int)(t2 % H2);
  int b  = (int)(t2 / H2);
  int blk = c4 / C, c = c4 % C;
  int dy = blk & 1, dx = blk >> 1;
  out[i] = in[(((long)b * H + 2 * y2 + dy) * W + 2 * x2 + dx) * C + c];
}

// ---------------------------------------------------------------- mean-pool + head
__global__ void k_pool_head(const float* __restrict__ a, const float* __restrict__ hw,
                            const float* __restrict__ hb, float* __restrict__ out,
                            int T, int C) {
  __shared__ float red[256];
  int b = blockIdx.x;
  float acc = 0.f;
  for (int c = threadIdx.x; c < C; c += blockDim.x) {
    float s = 0.f;
    for (int t = 0; t < T; ++t) s += a[((long)b * T + t) * C + c];
    acc += (s / T) * hw[c];
  }
  red[threadIdx.x] = acc;
  __syncthreads();
  for (int s = 128; s > 0; s >>= 1) {
    if ((int)threadIdx.x < s) red[threadIdx.x] += red[threadIdx.x + s];
    __syncthreads();
  }
  if (threadIdx.x == 0) out[b] = red[0] + hb[0];
}

// ---------------------------------------------------------------- host helpers
static inline void gemm(const __bf16* A, const __bf16* W, const float* bias,
                        float* OutF, __bf16* OutB, int M, int N, int K,
                        int act, int accum, hipStream_t s) {
  long tiles = (long)(M / 32) * (N / 32);
  int blocks = (int)((tiles + 7) / 8);
  k_gemm_bf16<<<blocks, 256, 0, s>>>(A, W, bias, OutF, OutB, M, N, K, act, accum);
}
static inline void layernorm(const float* in, const float* g, const float* b,
                             float* of, __bf16* ob, int M, int C, hipStream_t s) {
  k_layernorm<<<(M + 7) / 8, 256, 0, s>>>(in, g, b, of, ob, M, C);
}

extern "C" void kernel_launch(void* const* d_in, const int* in_sizes, int n_in,
                              void* d_out, int out_size, void* d_ws, size_t ws_size,
                              hipStream_t stream) {
  (void)in_sizes; (void)n_in; (void)out_size; (void)ws_size;
  int idx = 0;
  auto nxt = [&]() { return (const float*)d_in[idx++]; };

  const float* x_in    = nxt();
  const float* patch_w = nxt();
  const float* patch_b = nxt();
  const float* pn_g    = nxt();
  const float* pn_b    = nxt();

  struct Blk { const float *n1g,*n1b,*qkvw,*qkvb,*rpb,*projw,*projb,*n2g,*n2b,*f1w,*f1b,*f2w,*f2b; };
  Blk blocks[12];
  const float* mg[3]; const float* mb[3]; const float* mw[3];
  const int depths[4] = {2, 2, 6, 2};
  const int nheads[4] = {3, 6, 12, 24};
  int bi = 0;
  for (int s = 0; s < 4; ++s) {
    for (int j = 0; j < depths[s]; ++j) {
      Blk& B = blocks[bi++];
      B.n1g = nxt(); B.n1b = nxt();
      B.qkvw = nxt(); B.qkvb = nxt();
      B.rpb = nxt();
      B.projw = nxt(); B.projb = nxt();
      B.n2g = nxt(); B.n2b = nxt();
      B.f1w = nxt(); B.f1b = nxt();
      B.f2w = nxt(); B.f2b = nxt();
    }
    if (s < 3) { mg[s] = nxt(); mb[s] = nxt(); mw[s] = nxt(); }
  }
  const float* ng = nxt(); const float* nb = nxt();
  const float* hw = nxt(); const float* hb = nxt();

  const int Bn = 32;
  const long SZ = 100352L * 96;   // 9,633,792 elements
  float*  X     = (float*)d_ws;                 // residual stream fp32
  float*  Mg    = X + SZ;                       // merge-gather out fp32 / final LN fp32
  __bf16* Abf   = (__bf16*)(Mg + SZ);           // LN outputs bf16
  __bf16* Bbf   = Abf + SZ;                     // attention out bf16
  __bf16* BIGbf = Bbf + SZ;                     // qkv / fc1 out bf16 (4*SZ)
  __bf16* Wbf   = BIGbf + 4 * SZ;               // converted weights bf16

  // ---- convert all GEMM weights to bf16 once (deterministic, cheap) ----
  long woff = 0;
  auto cvt = [&](const float* src, long n) -> const __bf16* {
    __bf16* dst = Wbf + woff;
    woff += n;
    k_cvt_bf16<<<(int)((n + 255) / 256), 256, 0, stream>>>(src, dst, n);
    return dst;
  };
  const __bf16 *qkvwb[12], *projwb[12], *f1wb[12], *f2wb[12], *mwb[3];
  {
    int C = 96;
    bi = 0;
    for (int s = 0; s < 4; ++s) {
      for (int j = 0; j < depths[s]; ++j) {
        Blk& B = blocks[bi];
        qkvwb[bi] = cvt(B.qkvw, (long)3 * C * C);
        projwb[bi] = cvt(B.projw, (long)C * C);
        f1wb[bi] = cvt(B.f1w, (long)4 * C * C);
        f2wb[bi] = cvt(B.f2w, (long)4 * C * C);
        ++bi;
      }
      if (s < 3) { mwb[s] = cvt(mw[s], (long)8 * C * C); C *= 2; }
    }
  }

  // ---- patch embed + patch norm (fp32 residual stream) ----
  {
    long total = (long)Bn * 56 * 56 * 96;
    k_patch_embed<<<(int)((total + 255) / 256), 256, 0, stream>>>(x_in, patch_w, patch_b, X, Bn);
    layernorm(X, pn_g, pn_b, X, nullptr, Bn * 56 * 56, 96, stream);
  }

  int H = 56, W = 56, C = 96;
  bi = 0;
  for (int s = 0; s < 4; ++s) {
    for (int j = 0; j < depths[s]; ++j) {
      Blk& K = blocks[bi];
      int M = Bn * H * W;
      int shift = (j % 2 == 1 && H > 7) ? 3 : 0;

      layernorm(X, K.n1g, K.n1b, nullptr, Abf, M, C, stream);
      gemm(Abf, qkvwb[bi], K.qkvb, nullptr, BIGbf, M, 3 * C, C, 0, 0, stream);
      dim3 ag(Bn * (H / 7) * (W / 7), nheads[s]);
      k_attention<<<ag, 32, 0, stream>>>(BIGbf, Bbf, K.rpb, Bn, H, W, C, nheads[s], shift);
      gemm(Bbf, projwb[bi], K.projb, X, nullptr, M, C, C, 0, 1, stream);   // residual add
      layernorm(X, K.n2g, K.n2b, nullptr, Abf, M, C, stream);
      gemm(Abf, f1wb[bi], K.f1b, nullptr, BIGbf, M, 4 * C, C, 1, 0, stream);  // GELU
      gemm(BIGbf, f2wb[bi], K.f2b, X, nullptr, M, C, 4 * C, 0, 1, stream); // residual add
      ++bi;
    }
    if (s < 3) {
      long total = (long)Bn * (H / 2) * (W / 2) * 4 * C;
      k_merge_gather<<<(int)((total + 255) / 256), 256, 0, stream>>>(X, Mg, Bn, H, W, C);
      H /= 2; W /= 2;
      int M2 = Bn * H * W;
      layernorm(Mg, mg[s], mb[s], nullptr, Abf, M2, 4 * C, stream);
      gemm(Abf, mwb[s], nullptr, X, nullptr, M2, 2 * C, 4 * C, 0, 0, stream);
      C *= 2;
    }
  }

  layernorm(X, ng, nb, Mg, nullptr, Bn * H * W, C, stream);
  k_pool_head<<<Bn, 256, 0, stream>>>(Mg, hw, hb, (float*)d_out, H * W, C);
}
